// DDSPGenerator_65163243815756
// MI455X (gfx1250) — matrix operand order repaired and checked by reading, not compiled, not run on hardware
//
#include <hip/hip_runtime.h>
#include <hip/hip_bf16.h>
#include <math.h>

// ---------------------------------------------------------------------------
// DDSP generator for MI455X (gfx1250, wave32).
//  * conv1d -> implicit GEMM on v_wmma_f32_16x16x32_f16 (f16 in, f32 acc)
//    - activations: f16, [B][T+8][C] with 4-row zero halo  -> B-frag = 1 v16h load
//    - weights: pre-packed f16 in WMMA A-fragment order    -> A-frag = 1 v16h load
//    - 4 time-tiles per wave: A reuse across 4 WMMAs
//    - D store: 8 consecutive channels per lane -> 1 v8h (b128) store
//  * oscillator bank: per-(b,osc) LDS block scan + atomic accumulate
//  * noise bank: 32-point zero-phase circular convolution per frame
// ---------------------------------------------------------------------------

typedef __attribute__((ext_vector_type(16))) _Float16 v16h;
typedef __attribute__((ext_vector_type(8)))  _Float16 v8h;
typedef __attribute__((ext_vector_type(8)))  float    v8f;

#define ACT_LRELU  0
#define ACT_SQUARE 1
#define ACT_FREQ   2

#define BATCH     16
#define OUT_T     16384
#define N_OSC     128
#define SMALL_T   64
#define N_FRAMES  1024
#define N_COEFFS  17
#define PI_F      3.14159265358979323846f

__device__ __forceinline__ float lrelu_f(float v) { return v >= 0.f ? v : 0.2f * v; }

// ---------------------------------------------------------------------------
// Pack f32 weights [Cout][Cin][K] into f16 WMMA A-fragment order:
//   wp[(((kk*(Cin/32) + c)*coTiles + coTile)*32 + lane)*16 + j]
// lane half 0 -> K(cin) = {0..15}; half 1 -> {16..31} (contiguous, matches B-frag)
// ---------------------------------------------------------------------------
__global__ void pack_weights_kernel(const float* __restrict__ w, _Float16* __restrict__ wp,
                                    int Cin, int Cout, int K, int coTiles, int total)
{
    const int idx = blockIdx.x * blockDim.x + threadIdx.x;
    if (idx >= total) return;
    const int j      = idx & 15;
    const int lane   = (idx >> 4) & 31;
    const int rest   = idx >> 9;                 // (kk*cch + c)*coTiles + coTile
    const int coTile = rest % coTiles;
    const int rest2  = rest / coTiles;           // kk*cch + c
    const int cch    = Cin >> 5;
    const int c      = rest2 % cch;
    const int kk     = rest2 / cch;
    const int half   = lane >> 4;
    const int row    = (coTile << 4) + (lane & 15);
    const int kidx   = half * 16 + j;
    const float v = (row < Cout) ? w[((size_t)row * Cin + (c * 32 + kidx)) * K + kk] : 0.f;
    wp[idx] = (_Float16)v;
}

// x f32 [B][128][64]  ->  xp f16 [B][64+8][128], halo rows (4 each side) zeroed
__global__ void transpose_x_kernel(const float* __restrict__ x, _Float16* __restrict__ xp)
{
    const int idx = blockIdx.x * blockDim.x + threadIdx.x;
    const int total = BATCH * 72 * 128;
    if (idx >= total) return;
    const int c = idx & 127;
    const int q = (idx >> 7) % 72;
    const int b = idx / (72 * 128);
    const int t = q - 4;
    const float v = (t >= 0 && t < SMALL_T) ? x[((size_t)b * 128 + c) * SMALL_T + t] : 0.f;
    xp[idx] = (_Float16)v;
}

// zero the 8 halo time-rows of an f16 activation buffer [B][Tout+8][C]
__global__ void halo_zero_kernel(_Float16* __restrict__ buf, int Tout, int C)
{
    const int idx = blockIdx.x * blockDim.x + threadIdx.x;
    const int total = BATCH * 8 * C;
    if (idx >= total) return;
    const int c = idx % C;
    const int r = (idx / C) & 7;
    const int b = idx / (C * 8);
    const int q = (r < 4) ? r : Tout + r;        // rows 0..3 and Tout+4..Tout+7
    buf[((size_t)b * (Tout + 8) + q) * C + c] = (_Float16)0.f;
}

// ---------------------------------------------------------------------------
// conv1d as WMMA GEMM. One wave computes a 16(Cout) x 64(T) tile (4 WMMA N-tiles).
//   xp : f16 [B][Tin+8][Cin], valid data at row t+4, halos zero
//   wp : packed f16 A-fragments
// OUTF16: y = f16 [B][Tout+8][Cout] (+LReLU);  else: y = f32 [B][Cout][Tout] (+ACT)
// grid = (B, ceil(Cout/16), Tout/64), block = 32
// ---------------------------------------------------------------------------
template<int K, int ACT, bool UP2, bool OUTF16>
__global__ void conv1d_wmma_kernel(const _Float16* __restrict__ xp,
                                   const _Float16* __restrict__ wp,
                                   const float* __restrict__ bias,
                                   void* __restrict__ yraw,
                                   int Cin, int Cout, int Tout)
{
    constexpr int PAD = K / 2;
    const int b       = blockIdx.x;
    const int coTiles = gridDim.y;
    const int coTile  = blockIdx.y;
    const int co0     = coTile << 4;
    const int t0base  = blockIdx.z << 6;          // 4 tiles of 16
    const int lane    = threadIdx.x;              // wave32
    const int half    = lane >> 4;
    const int nCol    = lane & 15;
    const int Tin     = UP2 ? (Tout >> 1) : Tout;
    const int cch     = Cin >> 5;

    const _Float16* xb = xp + (size_t)b * (Tin + 8) * Cin;

    v8f acc0 = {}, acc1 = {}, acc2 = {}, acc3 = {};

    for (int kk = 0; kk < K; ++kk) {
        // B-fragment row pointers for the 4 time tiles (halo makes them always valid)
        const _Float16* xr0;
        const _Float16* xr1;
        const _Float16* xr2;
        const _Float16* xr3;
        {
            const int p0 = t0base +  0 + nCol + kk - PAD;
            const int p1 = t0base + 16 + nCol + kk - PAD;
            const int p2 = t0base + 32 + nCol + kk - PAD;
            const int p3 = t0base + 48 + nCol + kk - PAD;
            const int q0 = (UP2 ? (p0 >> 1) : p0) + 4;
            const int q1 = (UP2 ? (p1 >> 1) : p1) + 4;
            const int q2 = (UP2 ? (p2 >> 1) : p2) + 4;
            const int q3 = (UP2 ? (p3 >> 1) : p3) + 4;
            const int cofs = half << 4;
            xr0 = xb + (size_t)q0 * Cin + cofs;
            xr1 = xb + (size_t)q1 * Cin + cofs;
            xr2 = xb + (size_t)q2 * Cin + cofs;
            xr3 = xb + (size_t)q3 * Cin + cofs;
        }
        const _Float16* wr = wp + ((((size_t)kk * cch) * coTiles + coTile) * 32 + lane) * 16;
        const size_t wstride = (size_t)coTiles * 32 * 16;

        for (int c = 0; c < cch; ++c) {
            const v16h a  = *(const v16h*)wr;
            const v16h b0 = *(const v16h*)xr0;
            const v16h b1 = *(const v16h*)xr1;
            const v16h b2 = *(const v16h*)xr2;
            const v16h b3 = *(const v16h*)xr3;
            acc0 = __builtin_amdgcn_wmma_f32_16x16x32_f16(false, a, false, b0, (short)0, acc0, false, false);
            acc1 = __builtin_amdgcn_wmma_f32_16x16x32_f16(false, a, false, b1, (short)0, acc1, false, false);
            acc2 = __builtin_amdgcn_wmma_f32_16x16x32_f16(false, a, false, b2, (short)0, acc2, false, false);
            acc3 = __builtin_amdgcn_wmma_f32_16x16x32_f16(false, a, false, b3, (short)0, acc3, false, false);
            wr  += wstride;
            xr0 += 32; xr1 += 32; xr2 += 32; xr3 += 32;
        }
    }

    // ---- epilogue; acc[r] = (M = r + 8*half, N = nCol) per tile
    v8f accs[4] = {acc0, acc1, acc2, acc3};
    if (OUTF16) {
        _Float16* yp = (_Float16*)yraw;
#pragma unroll
        for (int i = 0; i < 4; ++i) {
            v8h o;
#pragma unroll
            for (int r = 0; r < 8; ++r) {
                float v = accs[i][r] + bias[co0 + (half << 3) + r];
                o[r] = (_Float16)lrelu_f(v);
            }
            const int t = t0base + i * 16 + nCol;
            *(v8h*)(yp + ((size_t)b * (Tout + 8) + t + 4) * Cout + co0 + (half << 3)) = o;
        }
    } else {
        float* yp = (float*)yraw;
#pragma unroll
        for (int i = 0; i < 4; ++i) {
            const int t = t0base + i * 16 + nCol;
#pragma unroll
            for (int r = 0; r < 8; ++r) {
                const int ch = co0 + (half << 3) + r;
                if (ch < Cout) {
                    float v = accs[i][r] + bias[ch];
                    if (ACT == ACT_SQUARE) {
                        v = v * v;
                    } else { // ACT_FREQ: starts + sigmoid(v)*diffs, geomspace(20,11025,128)
                        const float step  = logf(11025.f / 20.f) * (1.f / 127.f);
                        const float stop  = 20.f * expf(step * (float)ch);
                        const float start = (ch == 0) ? 0.f : 20.f * expf(step * (float)(ch - 1));
                        const float sig   = 1.f / (1.f + expf(-v));
                        v = start + sig * (stop - start);
                    }
                    yp[((size_t)b * Cout + ch) * Tout + t] = v;
                }
            }
        }
    }
}

// ---------------------------------------------------------------------------
// Oscillator bank: one 256-thread block per (b, osc) row. Fused x256 linear
// upsample (half-pixel, edge clamp), clip, LDS scan cumsum, sin*a, atomics.
// ---------------------------------------------------------------------------
__global__ void osc_bank_kernel(const float* __restrict__ fsm,
                                const float* __restrict__ lsm,
                                float* __restrict__ harm)
{
    __shared__ float fsh[SMALL_T];
    __shared__ float ash[SMALL_T];
    __shared__ float sdata[256];

    const int b   = blockIdx.x >> 7;
    const int osc = blockIdx.x & (N_OSC - 1);
    const int tid = threadIdx.x;

    const float* frow = fsm + ((size_t)b * N_OSC + osc) * SMALL_T;
    const float* arow = lsm + ((size_t)b * N_OSC + osc) * SMALL_T;
    if (tid < SMALL_T) { fsh[tid] = frow[tid]; ash[tid] = arow[tid]; }
    __syncthreads();

    const float ph_scale = 2.f * PI_F / 22050.f;
    float carry = 0.f;

    for (int chunk = 0; chunk < OUT_T / 256; ++chunk) {
        const int t = chunk * 256 + tid;
        const float pos = (t + 0.5f) * (1.f / 256.f) - 0.5f;
        const int   i0  = (int)floorf(pos);
        const float fr  = pos - (float)i0;
        const int   a0  = i0 < 0 ? 0 : i0;
        const int   a1  = (i0 + 1) > (SMALL_T - 1) ? (SMALL_T - 1) : (i0 + 1);
        float fv = fsh[a0] * (1.f - fr) + fsh[a1] * fr;
        const float av = ash[a0] * (1.f - fr) + ash[a1] * fr;
        fv = fminf(fmaxf(fv, 20.f), 11025.f);

        sdata[tid] = fv * ph_scale;
        __syncthreads();
        for (int off = 1; off < 256; off <<= 1) {
            const float tmp = (tid >= off) ? sdata[tid - off] : 0.f;
            __syncthreads();
            sdata[tid] += tmp;
            __syncthreads();
        }
        const float phase = carry + sdata[tid];
        const float total = sdata[255];
        atomicAdd(&harm[(size_t)b * OUT_T + t], sinf(phase) * av);
        carry += total;
        __syncthreads();
    }
}

// ---------------------------------------------------------------------------
// Noise bank: irfft(rfft(frame)*mags) == circular conv with zero-phase kernel
// H[d] = m0 + 2*sum_{k=1..15} m_k cos(2*pi*k*d/32) + m16 cos(pi*d), scale 1/32.
// One wave per (b, frame).
// ---------------------------------------------------------------------------
__global__ void noise_filt_kernel(const float* __restrict__ noise,
                                  const float* __restrict__ mags,
                                  float* __restrict__ filt)
{
    __shared__ float frame[32];
    __shared__ float H[32];
    __shared__ float msh[N_COEFFS];

    const int bi  = blockIdx.x;            // b * N_FRAMES + frame
    const int b   = bi >> 10;
    const int fi  = bi & (N_FRAMES - 1);
    const int tid = threadIdx.x;

    const int idx = fi * 16 + tid;         // hop=16, window=32, zero tail pad
    frame[tid] = (idx < OUT_T) ? noise[(size_t)b * OUT_T + idx] : 0.f;
    if (tid < N_COEFFS)
        msh[tid] = mags[((size_t)b * N_COEFFS + tid) * N_FRAMES + fi];
    __syncthreads();

    const float w32 = 2.f * PI_F / 32.f;
    float h = msh[0] + msh[16] * cosf(PI_F * (float)tid);
#pragma unroll
    for (int k = 1; k < 16; ++k)
        h += 2.f * msh[k] * cosf(w32 * (float)(k * tid));
    H[tid] = h;
    __syncthreads();

    float acc = 0.f;
#pragma unroll
    for (int n = 0; n < 32; ++n)
        acc += frame[n] * H[(tid - n) & 31];
    filt[(size_t)bi * 32 + tid] = acc * (1.f / 32.f);
}

// out = harmonic + 50%-overlap-add of filtered noise frames
__global__ void combine_kernel(const float* __restrict__ harm,
                               const float* __restrict__ filt,
                               float* __restrict__ out)
{
    const int idx = blockIdx.x * blockDim.x + threadIdx.x;
    if (idx >= BATCH * OUT_T) return;
    const int b  = idx >> 14;
    const int t  = idx & (OUT_T - 1);
    const int fr = t >> 4;
    const int r  = t & 15;
    float v = harm[idx];
    v += filt[(((size_t)b << 10) + fr) * 32 + r];
    if (t >= 16)
        v += filt[(((size_t)b << 10) + fr - 1) * 32 + 16 + r];
    out[idx] = v;
}

__global__ void fill_zero_kernel(float* __restrict__ p, int n)
{
    const int idx = blockIdx.x * blockDim.x + threadIdx.x;
    if (idx < n) p[idx] = 0.f;
}

// ---------------------------------------------------------------------------
extern "C" void kernel_launch(void* const* d_in, const int* in_sizes, int n_in,
                              void* d_out, int out_size, void* d_ws, size_t ws_size,
                              hipStream_t stream)
{
    (void)in_sizes; (void)n_in; (void)out_size; (void)ws_size;

    const float* x     = (const float*)d_in[0];
    const float* noise = (const float*)d_in[1];

    // ---- f16 workspace region -------------------------------------------------
    _Float16* h16 = (_Float16*)d_ws;
    size_t off16 = 0;
    auto alloc16 = [&](size_t n) { _Float16* p = h16 + off16; off16 += (n + 15) & ~(size_t)15; return p; };

    _Float16* act0 = alloc16((size_t)BATCH * 1032 * 512);   // up to T=1024 (+8 halo)
    _Float16* act1 = alloc16((size_t)BATCH * 1032 * 512);
    _Float16* xp   = alloc16((size_t)BATCH * 72 * 128);

    // pack one weight tensor -> WMMA A-fragment order
    auto pack = [&](int widx, int Cin, int Cout, int K) -> const _Float16* {
        const int coTiles = (Cout + 15) >> 4;
        const int total = K * (Cin >> 5) * coTiles * 32 * 16;
        _Float16* wpp = alloc16((size_t)total);
        pack_weights_kernel<<<(total + 255) / 256, 256, 0, stream>>>(
            (const float*)d_in[widx], wpp, Cin, Cout, K, coTiles, total);
        return wpp;
    };

    const _Float16* pmw0 = pack(2,  128, 512, 3);
    const _Float16* pmw1 = pack(4,  512, 512, 3);
    const _Float16* pmw2 = pack(6,  512, 512, 3);
    const _Float16* pmw3 = pack(8,  512, 512, 3);
    const _Float16* pfw  = pack(10, 512, 128, 3);
    const _Float16* pgw0 = pack(12, 128, 512, 3);
    const _Float16* pgw1 = pack(14, 512, 512, 3);
    const _Float16* pgw2 = pack(16, 512, 512, 3);
    const _Float16* pgw3 = pack(18, 512, 512, 3);
    const _Float16* plw  = pack(20, 512, 128, 3);
    const _Float16* pnw0 = pack(22, 128, 512, 7);
    const _Float16* pnw1 = pack(24, 512, 512, 7);
    const _Float16* pnw2 = pack(26, 512, 512, 7);
    const _Float16* pnw3 = pack(28, 512, 512, 7);
    const _Float16* pnlw = pack(30, 512,  17, 3);

    // ---- f32 workspace region -------------------------------------------------
    float* f32b = (float*)(h16 + off16);
    float* fsm  = f32b;                      // [16][128][64]
    float* lsm  = fsm + 131072;              // [16][128][64]
    float* nl   = lsm + 131072;              // [16][17][1024]
    float* harm = nl + 278528;               // [16][16384]
    float* filt = harm + 262144;             // [16][1024][32]

    const dim3 wave(32);
    auto B = [&](int i) { return (const float*)d_in[i]; };

    // input transpose + halo, harmonic accumulator zero
    transpose_x_kernel<<<(BATCH * 72 * 128 + 255) / 256, 256, 0, stream>>>(x, xp);
    fill_zero_kernel<<<(BATCH * OUT_T + 255) / 256, 256, 0, stream>>>(harm, BATCH * OUT_T);

    const int hz = (BATCH * 8 * 512 + 255) / 256;   // halo-zero launch size (C=512)

    // ---- frequency path (T=64, k=3) ----
    halo_zero_kernel<<<hz, 256, 0, stream>>>(act0, 64, 512);
    conv1d_wmma_kernel<3, ACT_LRELU, false, true ><<<dim3(16, 32, 1), wave, 0, stream>>>(xp,   pmw0, B(3),  act0, 128, 512, 64);
    halo_zero_kernel<<<hz, 256, 0, stream>>>(act1, 64, 512);
    conv1d_wmma_kernel<3, ACT_LRELU, false, true ><<<dim3(16, 32, 1), wave, 0, stream>>>(act0, pmw1, B(5),  act1, 512, 512, 64);
    halo_zero_kernel<<<hz, 256, 0, stream>>>(act0, 64, 512);
    conv1d_wmma_kernel<3, ACT_LRELU, false, true ><<<dim3(16, 32, 1), wave, 0, stream>>>(act1, pmw2, B(7),  act0, 512, 512, 64);
    halo_zero_kernel<<<hz, 256, 0, stream>>>(act1, 64, 512);
    conv1d_wmma_kernel<3, ACT_LRELU, false, true ><<<dim3(16, 32, 1), wave, 0, stream>>>(act0, pmw3, B(9),  act1, 512, 512, 64);
    conv1d_wmma_kernel<3, ACT_FREQ,  false, false><<<dim3(16,  8, 1), wave, 0, stream>>>(act1, pfw,  B(11), fsm,  512, 128, 64);

    // ---- loudness path (T=64, k=3) ----
    halo_zero_kernel<<<hz, 256, 0, stream>>>(act0, 64, 512);
    conv1d_wmma_kernel<3, ACT_LRELU, false, true ><<<dim3(16, 32, 1), wave, 0, stream>>>(xp,   pgw0, B(13), act0, 128, 512, 64);
    halo_zero_kernel<<<hz, 256, 0, stream>>>(act1, 64, 512);
    conv1d_wmma_kernel<3, ACT_LRELU, false, true ><<<dim3(16, 32, 1), wave, 0, stream>>>(act0, pgw1, B(15), act1, 512, 512, 64);
    halo_zero_kernel<<<hz, 256, 0, stream>>>(act0, 64, 512);
    conv1d_wmma_kernel<3, ACT_LRELU, false, true ><<<dim3(16, 32, 1), wave, 0, stream>>>(act1, pgw2, B(17), act0, 512, 512, 64);
    halo_zero_kernel<<<hz, 256, 0, stream>>>(act1, 64, 512);
    conv1d_wmma_kernel<3, ACT_LRELU, false, true ><<<dim3(16, 32, 1), wave, 0, stream>>>(act0, pgw3, B(19), act1, 512, 512, 64);
    conv1d_wmma_kernel<3, ACT_SQUARE, false, false><<<dim3(16, 8, 1), wave, 0, stream>>>(act1, plw,  B(21), lsm,  512, 128, 64);

    // ---- noise-envelope path (x2 upsample folded into gather; k=7) ----
    halo_zero_kernel<<<hz, 256, 0, stream>>>(act0, 128, 512);
    conv1d_wmma_kernel<7, ACT_LRELU, true,  true ><<<dim3(16, 32,  2), wave, 0, stream>>>(xp,   pnw0, B(23), act0, 128, 512, 128);
    halo_zero_kernel<<<hz, 256, 0, stream>>>(act1, 256, 512);
    conv1d_wmma_kernel<7, ACT_LRELU, true,  true ><<<dim3(16, 32,  4), wave, 0, stream>>>(act0, pnw1, B(25), act1, 512, 512, 256);
    halo_zero_kernel<<<hz, 256, 0, stream>>>(act0, 512, 512);
    conv1d_wmma_kernel<7, ACT_LRELU, true,  true ><<<dim3(16, 32,  8), wave, 0, stream>>>(act1, pnw2, B(27), act0, 512, 512, 512);
    halo_zero_kernel<<<hz, 256, 0, stream>>>(act1, 1024, 512);
    conv1d_wmma_kernel<7, ACT_LRELU, true,  true ><<<dim3(16, 32, 16), wave, 0, stream>>>(act0, pnw3, B(29), act1, 512, 512, 1024);
    conv1d_wmma_kernel<3, ACT_SQUARE, false, false><<<dim3(16, 2, 16), wave, 0, stream>>>(act1, pnlw, B(31), nl,   512,  17, 1024);

    // ---- oscillator bank (harmonic part) ----
    osc_bank_kernel<<<dim3(BATCH * N_OSC), 256, 0, stream>>>(fsm, lsm, harm);

    // ---- filtered-noise frames + combine ----
    noise_filt_kernel<<<dim3(BATCH * N_FRAMES), 32, 0, stream>>>(noise, nl, filt);
    combine_kernel<<<(BATCH * OUT_T + 255) / 256, 256, 0, stream>>>(harm, filt, (float*)d_out);
}
